// Speller_18030272709218
// MI455X (gfx1250) — matrix operand order repaired and checked by reading, not compiled, not run on hardware
//
#include <hip/hip_runtime.h>
#include <hip/hip_bf16.h>

// ---------------------------------------------------------------------------
// LAS Speller decoder, persistent-kernel implementation for gfx1250 (MI455X).
// 128 sequential steps; per step: gates GEMM (bf16 WMMA) + LSTM cell fused,
// attention (energy/softmax/ctx) fused per-batch-row in one block, output
// GEMM (bf16 WMMA) + log-softmax + greedy one-hot feedback.
// Grid-wide sync: atomic counter/generation barrier (3 syncs per step).
// Branch-free sigmoid/tanh (__expf + v_rcp_f32) keep the recurrence chain
// free of exec-mask divergence.
// ---------------------------------------------------------------------------

typedef __attribute__((ext_vector_type(16))) __bf16 v16bf;
typedef __attribute__((ext_vector_type(8)))  float  v8f;
typedef __attribute__((ext_vector_type(2)))  __bf16 bf16x2;

#define NBLK 64
#define NTHR 256

#define Bx  32
#define Tt  1024
#define Dd  512
#define Hh  512
#define Vv  64
#define Ss  128
#define XW  576            // V + H  (width of x = [onehot | ctx])
#define KIN 1088           // XW + H (combined input [onehot | ctx | h])
#define G4  2048           // 4*H

// -------------------------- grid-wide barrier ------------------------------
__device__ __forceinline__ void grid_sync(unsigned* cnt, unsigned* gen) {
  __syncthreads();
  if (threadIdx.x == 0) {
    unsigned g = __hip_atomic_load(gen, __ATOMIC_RELAXED, __HIP_MEMORY_SCOPE_AGENT);
    unsigned a = __hip_atomic_fetch_add(cnt, 1u, __ATOMIC_ACQ_REL, __HIP_MEMORY_SCOPE_AGENT);
    if (a == (unsigned)(NBLK - 1)) {
      __hip_atomic_store(cnt, 0u, __ATOMIC_RELAXED, __HIP_MEMORY_SCOPE_AGENT);
      __hip_atomic_store(gen, g + 1u, __ATOMIC_RELEASE, __HIP_MEMORY_SCOPE_AGENT);
    } else {
      while (__hip_atomic_load(gen, __ATOMIC_ACQUIRE, __HIP_MEMORY_SCOPE_AGENT) == g) {
        __builtin_amdgcn_s_sleep(1);
      }
    }
  }
  __syncthreads();
}

// ----------------- WMMA operand loaders (CDNA5 VGPR layouts) ---------------
// A-matrix 16x32 bf16 (M x K), row-major source, row stride ld.
__device__ __forceinline__ v16bf load_a16x32(const __bf16* __restrict__ base,
                                             int ld, int rowBase, int kBase, int lane) {
  const int m  = rowBase + (lane & 15);
  const int hi = lane >> 4;
  const __bf16* p = base + (size_t)m * ld + kBase + 8 * hi;
  v16bf a;
#pragma unroll
  for (int i = 0; i < 4; ++i) {
    bf16x2 lo = *(const bf16x2*)(p + 2 * i);        // K = 8*hi + 2i, +1
    bf16x2 up = *(const bf16x2*)(p + 16 + 2 * i);   // K = 16 + 8*hi + 2i, +1
    a[2 * i] = lo.x; a[2 * i + 1] = lo.y;
    a[8 + 2 * i] = up.x; a[9 + 2 * i] = up.y;
  }
  return a;
}

// B-matrix 32x16 bf16 (K x N) taken as column n = row n of W[N][K] (row-major).
__device__ __forceinline__ v16bf load_bT32x16(const __bf16* __restrict__ W,
                                              int ld, int nBase, int kBase, int lane) {
  const int n  = nBase + (lane & 15);
  const int hi = lane >> 4;
  const __bf16* p = W + (size_t)n * ld + kBase + 8 * hi;
  v16bf b;
#pragma unroll
  for (int i = 0; i < 4; ++i) {
    bf16x2 v = *(const bf16x2*)(p + 2 * i);
    bf16x2 u = *(const bf16x2*)(p + 16 + 2 * i);
    b[2 * i] = v.x; b[2 * i + 1] = v.y;
    b[8 + 2 * i] = u.x; b[9 + 2 * i] = u.y;
  }
  return b;
}

__device__ __forceinline__ v8f wmma_bf16(v16bf a, v16bf b, v8f c) {
  return __builtin_amdgcn_wmma_f32_16x16x32_bf16(false, a, false, b, (short)0, c,
                                                 false, false);
}

// Branch-free activations: clamped __expf + v_rcp_f32. No exec-mask splits.
__device__ __forceinline__ float sigmoid_fast(float x) {
  const float xc = fminf(fmaxf(x, -30.f), 30.f);
  return __builtin_amdgcn_rcpf(1.f + __expf(-xc));
}
__device__ __forceinline__ float tanh_fast(float x) {
  const float xc = fminf(fmaxf(x, -15.f), 15.f);
  const float t = __expf(2.f * xc);
  return (t - 1.f) * __builtin_amdgcn_rcpf(t + 1.f);
}

// ------------------------------ init kernel --------------------------------
__global__ void speller_sync_init(unsigned* s) {
  if (threadIdx.x < 2) s[threadIdx.x] = 0u;
}

// --------------------------- persistent kernel -----------------------------
__global__ __launch_bounds__(NTHR) void speller_persistent(
    const float* __restrict__ LF,
    const float* __restrict__ W_ih, const float* __restrict__ W_hh,
    const float* __restrict__ b_ih, const float* __restrict__ b_hh,
    const float* __restrict__ W_out, const float* __restrict__ b_out,
    float* __restrict__ out_pred,   // [S,B,V]
    float* __restrict__ out_attn,   // [S,B,T]
    unsigned* __restrict__ sync_,
    __bf16* __restrict__ xh,        // [B,KIN] bf16 recurrent input [ow|ctx|h]
    __bf16* __restrict__ Wc,        // [4H,KIN] bf16 combined [W_ih|W_hh]
    __bf16* __restrict__ Wob,       // [V,2H]  bf16 W_out
    float* __restrict__ Hf,         // [B,H] f32 hidden state
    float* __restrict__ plog)       // [B,V] partial logits (h-half + bias)
{
  const int tid  = threadIdx.x;
  const int gid  = blockIdx.x * NTHR + tid;
  const int lane = tid & 31;
  unsigned* cnt = sync_;
  unsigned* gen = sync_ + 1;

  __shared__ float s_buf[2048];   // P2: energies/exp (1024) ; P3: logits (2048)
  __shared__ float s_red[16];

  // ---------------- one-time prep (runs each launch; deterministic) --------
  for (int idx = gid; idx < G4 * KIN; idx += NBLK * NTHR) {
    const int g = idx / KIN, k = idx - g * KIN;
    const float v = (k < XW) ? W_ih[(size_t)g * XW + k]
                             : W_hh[(size_t)g * Hh + (k - XW)];
    Wc[idx] = (__bf16)v;
  }
  for (int idx = gid; idx < Vv * 2 * Hh; idx += NBLK * NTHR)
    Wob[idx] = (__bf16)W_out[idx];
  for (int idx = gid; idx < Bx * Hh; idx += NBLK * NTHR) Hf[idx] = 0.f;
  for (int idx = gid; idx < Bx * KIN; idx += NBLK * NTHR) {
    const int b = idx / KIN, k = idx - b * KIN;
    float v;
    if (k < Vv)        v = (k == 0) ? 1.f : 0.f;              // ow0 = onehot(0)
    else if (k < XW)   v = LF[(size_t)b * Tt * Dd + (k - Vv)]; // LF[:,0,:]
    else               v = 0.f;                                // h0 = 0
    xh[idx] = (__bf16)v;
  }

  // P1 wave-constant state: tile ownership, biases, cell state (in VGPRs).
  const int w_p1  = blockIdx.x * 8 + (tid >> 5);  // 0..63 for blocks 0..7
  const int mt_p1 = w_p1 >> 5;                    // batch tile 0..1
  const int hj_p1 = w_p1 & 31;                    // hidden chunk 0..31
  const int j_p1  = hj_p1 * 16 + (lane & 15);
  float bi_ = 0.f, bf_ = 0.f, bg_ = 0.f, bo_ = 0.f;
  if (blockIdx.x < 8) {
    bi_ = b_ih[j_p1]          + b_hh[j_p1];
    bf_ = b_ih[Hh + j_p1]     + b_hh[Hh + j_p1];
    bg_ = b_ih[2 * Hh + j_p1] + b_hh[2 * Hh + j_p1];
    bo_ = b_ih[3 * Hh + j_p1] + b_hh[3 * Hh + j_p1];
  }
  float c_reg[8] = {0.f, 0.f, 0.f, 0.f, 0.f, 0.f, 0.f, 0.f};

  grid_sync(cnt, gen);

  for (int s = 0; s < Ss; ++s) {
    // ================= P1: gates GEMM + LSTM cell (blocks 0..7) ============
    if (blockIdx.x < 8) {
      v8f ai, af, ag, ao;                          // bias folded into C init
#pragma unroll
      for (int r = 0; r < 8; ++r) { ai[r] = bi_; af[r] = bf_; ag[r] = bg_; ao[r] = bo_; }
      for (int kt = 0; kt < KIN / 32; ++kt) {
        const v16bf a  = load_a16x32(xh, KIN, mt_p1 * 16, kt * 32, lane);
        const v16bf b0 = load_bT32x16(Wc, KIN, (hj_p1)      * 16, kt * 32, lane);
        ai = wmma_bf16(a, b0, ai);
        const v16bf b1 = load_bT32x16(Wc, KIN, (32 + hj_p1) * 16, kt * 32, lane);
        af = wmma_bf16(a, b1, af);
        const v16bf b2 = load_bT32x16(Wc, KIN, (64 + hj_p1) * 16, kt * 32, lane);
        ag = wmma_bf16(a, b2, ag);
        const v16bf b3 = load_bT32x16(Wc, KIN, (96 + hj_p1) * 16, kt * 32, lane);
        ao = wmma_bf16(a, b3, ao);
      }
      const int hi = lane >> 4;
#pragma unroll
      for (int r = 0; r < 8; ++r) {
        const int bb = mt_p1 * 16 + r + 8 * hi;    // batch row (D-layout)
        const float ii = sigmoid_fast(ai[r]);
        const float ff = sigmoid_fast(af[r]);
        const float gg = tanh_fast(ag[r]);
        const float oo = sigmoid_fast(ao[r]);
        const float c  = ff * c_reg[r] + ii * gg;  // cell state lives in VGPRs
        c_reg[r] = c;
        const float h = oo * tanh_fast(c);
        Hf[bb * Hh + j_p1] = h;
        xh[(size_t)bb * KIN + XW + j_p1] = (__bf16)h;
      }
    }
    grid_sync(cnt, gen);

    // ====== P2: energy -> softmax -> ctx (blocks 0..31, one per batch) =====
    //        block 32 concurrently: h-half of output logits via WMMA
    if (blockIdx.x < Bx) {
      const int b  = blockIdx.x;
      const int wv = tid >> 5;
      float hr[16];
      {
        const float4* hp = (const float4*)(Hf + b * Hh + lane * 16);
#pragma unroll
        for (int q = 0; q < 4; ++q) {
          const float4 v = hp[q];
          hr[4 * q] = v.x; hr[4 * q + 1] = v.y; hr[4 * q + 2] = v.z; hr[4 * q + 3] = v.w;
        }
      }
      for (int tt = 0; tt < 128; ++tt) {           // energy[b][t] = h . LF[b,t,:]
        const int t = wv * 128 + tt;
        const float4* lp = (const float4*)(LF + ((size_t)b * Tt + t) * Dd + lane * 16);
        float e = 0.f;
#pragma unroll
        for (int q = 0; q < 4; ++q) {
          const float4 v = lp[q];
          e = fmaf(v.x, hr[4 * q], e);     e = fmaf(v.y, hr[4 * q + 1], e);
          e = fmaf(v.z, hr[4 * q + 2], e); e = fmaf(v.w, hr[4 * q + 3], e);
        }
#pragma unroll
        for (int m = 16; m; m >>= 1) e += __shfl_xor(e, m, 32);
        if (lane == 0) s_buf[t] = e;
      }
      __syncthreads();
      // block-local softmax over T=1024
      float lm = -3.4e38f;
#pragma unroll
      for (int q = 0; q < 4; ++q) lm = fmaxf(lm, s_buf[tid + 256 * q]);
#pragma unroll
      for (int m = 16; m; m >>= 1) lm = fmaxf(lm, __shfl_xor(lm, m, 32));
      if (lane == 0) s_red[wv] = lm;
      __syncthreads();
      float mx = s_red[0];
#pragma unroll
      for (int q = 1; q < 8; ++q) mx = fmaxf(mx, s_red[q]);
      float ls = 0.f;
#pragma unroll
      for (int q = 0; q < 4; ++q) {
        const float ex = __expf(s_buf[tid + 256 * q] - mx);
        s_buf[tid + 256 * q] = ex;                 // own slots only: no race
        ls += ex;
      }
#pragma unroll
      for (int m = 16; m; m >>= 1) ls += __shfl_xor(ls, m, 32);
      if (lane == 0) s_red[8 + wv] = ls;
      __syncthreads();
      float sum = s_red[8];
#pragma unroll
      for (int q = 1; q < 8; ++q) sum += s_red[8 + q];
      const float inv = __builtin_amdgcn_rcpf(sum);
      {
        float* ap = out_attn + ((size_t)s * Bx + b) * Tt;
#pragma unroll
        for (int q = 0; q < 4; ++q) ap[tid + 256 * q] = s_buf[tid + 256 * q] * inv;
      }
      // ctx[b][d] = inv * sum_t exp_e[t] * LF[b,t,d]
      const int d0 = tid * 2;
      float ax = 0.f, ay = 0.f;
      const float* lfb = LF + (size_t)b * Tt * Dd + d0;
#pragma unroll 4
      for (int t = 0; t < Tt; ++t) {
        const float a = s_buf[t];
        const float2 v = *(const float2*)(lfb + (size_t)t * Dd);
        ax = fmaf(a, v.x, ax);
        ay = fmaf(a, v.y, ay);
      }
      ax *= inv; ay *= inv;
      xh[(size_t)b * KIN + Vv + d0]     = (__bf16)ax;
      xh[(size_t)b * KIN + Vv + d0 + 1] = (__bf16)ay;
    } else if (blockIdx.x == Bx) {
      // partial logits: b_out + h @ W_out[:, :H]^T   (M=32, N=64, K=512)
      const int wv = tid >> 5;
      const int mt = wv >> 2, nt = wv & 3;
      const int n  = nt * 16 + (lane & 15);
      const int hi = lane >> 4;
      v8f acc;
      const float bo = b_out[n];
#pragma unroll
      for (int r = 0; r < 8; ++r) acc[r] = bo;
      for (int kt = 0; kt < Hh / 32; ++kt) {
        const v16bf a  = load_a16x32(xh, KIN, mt * 16, XW + kt * 32, lane);
        const v16bf bb = load_bT32x16(Wob, 2 * Hh, nt * 16, kt * 32, lane);
        acc = wmma_bf16(a, bb, acc);
      }
#pragma unroll
      for (int r = 0; r < 8; ++r)
        plog[(mt * 16 + r + 8 * hi) * Vv + n] = acc[r];
    }
    grid_sync(cnt, gen);

    // ===== P3 (block 0): ctx-half of logits + log-softmax + argmax =========
    if (blockIdx.x == 0) {
      const int wv = tid >> 5;
      const int mt = wv >> 2, nt = wv & 3;
      const int n  = nt * 16 + (lane & 15);
      const int hi = lane >> 4;
      v8f acc;
#pragma unroll
      for (int r = 0; r < 8; ++r)
        acc[r] = plog[(mt * 16 + r + 8 * hi) * Vv + n];
      for (int kt = 0; kt < Hh / 32; ++kt) {
        const v16bf a  = load_a16x32(xh, KIN, mt * 16, Vv + kt * 32, lane);
        const v16bf bb = load_bT32x16(Wob, 2 * Hh, nt * 16, Hh + kt * 32, lane);
        acc = wmma_bf16(a, bb, acc);
      }
#pragma unroll
      for (int r = 0; r < 8; ++r)
        s_buf[(mt * 16 + r + 8 * hi) * Vv + n] = acc[r];
      __syncthreads();
      // 8 waves x 4 batch rows: log-softmax + first-index argmax + one-hot
      for (int jb = 0; jb < 4; ++jb) {
        const int b  = wv * 4 + jb;
        const float v0 = s_buf[b * Vv + lane];
        const float v1 = s_buf[b * Vv + 32 + lane];
        float mx = fmaxf(v0, v1);
#pragma unroll
        for (int m = 16; m; m >>= 1) mx = fmaxf(mx, __shfl_xor(mx, m, 32));
        float sm = __expf(v0 - mx) + __expf(v1 - mx);
#pragma unroll
        for (int m = 16; m; m >>= 1) sm += __shfl_xor(sm, m, 32);
        const float lz = mx + __logf(sm);
        float* pp = out_pred + ((size_t)s * Bx + b) * Vv;
        pp[lane]      = v0 - lz;
        pp[lane + 32] = v1 - lz;
        float bv; int bi2;
        if (v1 > v0) { bv = v1; bi2 = lane + 32; } else { bv = v0; bi2 = lane; }
#pragma unroll
        for (int m = 16; m; m >>= 1) {
          const float ov = __shfl_xor(bv, m, 32);
          const int   oi = __shfl_xor(bi2, m, 32);
          if (ov > bv || (ov == bv && oi < bi2)) { bv = ov; bi2 = oi; }
        }
        xh[(size_t)b * KIN + lane]      = (__bf16)((lane == bi2) ? 1.f : 0.f);
        xh[(size_t)b * KIN + 32 + lane] = (__bf16)((lane + 32 == bi2) ? 1.f : 0.f);
      }
    }
    grid_sync(cnt, gen);
  }
}

// ------------------------------ host launch --------------------------------
extern "C" void kernel_launch(void* const* d_in, const int* in_sizes, int n_in,
                              void* d_out, int out_size, void* d_ws, size_t ws_size,
                              hipStream_t stream) {
  (void)in_sizes; (void)n_in; (void)out_size; (void)ws_size;
  const float* LF    = (const float*)d_in[0];
  const float* W_ih  = (const float*)d_in[1];
  const float* W_hh  = (const float*)d_in[2];
  const float* b_ih  = (const float*)d_in[3];
  const float* b_hh  = (const float*)d_in[4];
  const float* W_out = (const float*)d_in[5];
  const float* b_out = (const float*)d_in[6];

  float* out_pred = (float*)d_out;                         // [S,B,V]
  float* out_attn = out_pred + (size_t)Ss * Bx * Vv;       // [S,B,T]

  char* ws = (char*)d_ws;
  size_t off = 0;
  auto take = [&](size_t bytes) -> size_t {
    size_t o = off;
    off += (bytes + 255) & ~(size_t)255;
    return o;
  };
  unsigned* sync_ = (unsigned*)(ws + take(2 * sizeof(unsigned)));
  __bf16* xh  = (__bf16*)(ws + take((size_t)Bx * KIN * 2));
  __bf16* Wc  = (__bf16*)(ws + take((size_t)G4 * KIN * 2));
  __bf16* Wob = (__bf16*)(ws + take((size_t)Vv * 2 * Hh * 2));
  float*  Hf   = (float*)(ws + take((size_t)Bx * Hh * 4));
  float*  plog = (float*)(ws + take((size_t)Bx * Vv * 4));

  speller_sync_init<<<1, 64, 0, stream>>>(sync_);
  speller_persistent<<<NBLK, NTHR, 0, stream>>>(
      LF, W_ih, W_hh, b_ih, b_hh, W_out, b_out,
      out_pred, out_attn, sync_, xh, Wc, Wob, Hf, plog);
}
